// MoE_42417097015419
// MI455X (gfx1250) — matrix operand order repaired and checked by reading, not compiled, not run on hardware
//
#include <hip/hip_runtime.h>
#include <math.h>

typedef __attribute__((ext_vector_type(16))) __bf16 v16bf;
typedef __attribute__((ext_vector_type(8)))  __bf16 v8bf;
typedef __attribute__((ext_vector_type(4)))  __bf16 v4bf;
typedef __attribute__((ext_vector_type(8)))  float  v8f;
typedef __attribute__((ext_vector_type(4)))  int    v4i;

constexpr int Bt = 8192;   // tokens
constexpr int Dd = 1024;   // model dim
constexpr int Hh = 2048;   // hidden dim
constexpr int Ee = 8;      // experts
constexpr int BM = 128;    // token tile
constexpr int BN = 128;    // output tile
constexpr int BK = 32;     // k tile (bf16 wmma k=32)
constexpr int LDK = BK + 8; // padded LDS row stride (keeps 16B chunks aligned)

#if __has_builtin(__builtin_amdgcn_global_load_async_to_lds_b128)
#define HAVE_ASYNC 1
#else
#define HAVE_ASYNC 0
#endif

#if HAVE_ASYNC
typedef __attribute__((address_space(1))) v4i gv4i_t;  // global int4*
typedef __attribute__((address_space(3))) v4i lv4i_t;  // LDS int4*
__device__ __forceinline__ void wait_async0() {
#if __has_builtin(__builtin_amdgcn_s_wait_asynccnt)
  __builtin_amdgcn_s_wait_asynccnt(0);
#else
  asm volatile("s_wait_asynccnt 0x0" ::: "memory");
#endif
}
#endif

// ---------------------------------------------------------------------------
// Fragment loaders (LDS tiles stored [row][LDK] bf16, 16B-aligned chunks)
// A 16x32 bf16: lane<16 row=l, K = 0..7 & 16..23 ; lane>=16 row=l, K=8..15 & 24..31
// B 32x16 bf16: lane<16 col=l, K = 0..7 & 16..23 ; lane>=16 col=l, K=8..15 & 24..31
// ---------------------------------------------------------------------------
__device__ __forceinline__ v16bf frag_cat(v8bf lo, v8bf hi) {
  v16bf r;
#pragma unroll
  for (int i = 0; i < 8; ++i) { r[i] = lo[i]; r[8 + i] = hi[i]; }
  return r;
}

__device__ __forceinline__ v16bf load_afrag(const __bf16* base, int row0, int lane) {
  int l  = lane & 15;
  int k0 = (lane >> 4) * 8;
  const __bf16* p = base + (row0 + l) * LDK;
  v8bf lo = *(const v8bf*)(p + k0);
  v8bf hi = *(const v8bf*)(p + 16 + k0);
  return frag_cat(lo, hi);
}

__device__ __forceinline__ v16bf load_bfrag(const __bf16* base, int col0, int lane) {
  int l  = lane & 15;
  int k0 = (lane >> 4) * 8;
  const __bf16* p = base + (col0 + l) * LDK + k0;
  v8bf lo = *(const v8bf*)(p);
  v8bf hi = *(const v8bf*)(p + 16);
  return frag_cat(lo, hi);
}

// ---------------------------------------------------------------------------
// 1) Gating: one wave per token. logits = x[b,:] @ w_gate ; top-2 softmax ;
//    compact (expert, rank) lists via atomics ; importance accumulation.
// ---------------------------------------------------------------------------
__global__ __launch_bounds__(256) void gate_kernel(
    const float* __restrict__ x, const float* __restrict__ wg,
    int* __restrict__ counts, float* __restrict__ importance,
    int* __restrict__ token_list, float* __restrict__ gate_vals) {
  int gwave = (blockIdx.x * blockDim.x + threadIdx.x) >> 5;
  int lane  = threadIdx.x & 31;
  if (gwave >= Bt) return;

  const float* xr = x + (size_t)gwave * Dd;
  float acc[Ee] = {0.f, 0.f, 0.f, 0.f, 0.f, 0.f, 0.f, 0.f};
  for (int d = lane; d < Dd; d += 32) {
    float xv = xr[d];
    const float* g = wg + d * Ee;
#pragma unroll
    for (int e = 0; e < Ee; ++e) acc[e] += xv * g[e];
  }
#pragma unroll
  for (int off = 16; off >= 1; off >>= 1) {
#pragma unroll
    for (int e = 0; e < Ee; ++e) acc[e] += __shfl_xor(acc[e], off, 32);
  }

  if (lane == 0) {
    int i0 = 0; float l0 = acc[0];
#pragma unroll
    for (int e = 1; e < Ee; ++e) if (acc[e] > l0) { l0 = acc[e]; i0 = e; }
    int i1 = -1; float l1 = -3.4e38f;
#pragma unroll
    for (int e = 0; e < Ee; ++e) if (e != i0 && acc[e] > l1) { l1 = acc[e]; i1 = e; }
    float e1 = __expf(l1 - l0);
    float inv = 1.0f / (1.0f + e1);
    float w0 = inv, w1 = e1 * inv;

    int r0 = atomicAdd(&counts[i0], 1);
    token_list[i0 * Bt + r0] = gwave;
    gate_vals[i0 * Bt + r0] = w0;
    atomicAdd(&importance[i0], w0);

    int r1 = atomicAdd(&counts[i1], 1);
    token_list[i1 * Bt + r1] = gwave;
    gate_vals[i1 * Bt + r1] = w1;
    atomicAdd(&importance[i1], w1);
  }
}

// ---------------------------------------------------------------------------
// 2) Offsets scan + cv^2 load-balancing loss (ddof=1), loss -> out[B*D]
// ---------------------------------------------------------------------------
__global__ void finalize_kernel(const int* __restrict__ counts,
                                const float* __restrict__ importance,
                                int* __restrict__ offsets,
                                float* __restrict__ out_loss) {
  if (threadIdx.x != 0 || blockIdx.x != 0) return;
  int off = 0;
  float mi = 0.f, ml = 0.f;
  for (int e = 0; e < Ee; ++e) {
    offsets[e] = off;
    off += counts[e];
    mi += importance[e];
    ml += (float)counts[e];
  }
  mi *= (1.0f / Ee); ml *= (1.0f / Ee);
  float vi = 0.f, vl = 0.f;
  for (int e = 0; e < Ee; ++e) {
    float di = importance[e] - mi; vi += di * di;
    float dl = (float)counts[e] - ml; vl += dl * dl;
  }
  vi *= (1.0f / (Ee - 1)); vl *= (1.0f / (Ee - 1));
  *out_loss = 1e-2f * (vi / (mi * mi + 1e-10f) + vl / (ml * ml + 1e-10f));
}

// ---------------------------------------------------------------------------
// 3) fc1: h[off_e+rank, n] = gelu( gather(x) @ W1[e] + b1[e] )  (bf16 out)
//    Block: 128 tokens x 128 H cols, 8 waves, each 64x32 (4x2 wmma accs).
//    Double-buffered LDS pipeline: global loads (k+1) || WMMA (k).
// ---------------------------------------------------------------------------
__global__ __launch_bounds__(256) void fc1_kernel(
    const float* __restrict__ x, const float* __restrict__ W1,
    const float* __restrict__ b1, const int* __restrict__ counts,
    const int* __restrict__ offsets, const int* __restrict__ token_list,
    __bf16* __restrict__ h) {
  int e  = blockIdx.z;
  int t0 = blockIdx.y * BM;
  int n0 = blockIdx.x * BN;
  int cnt = counts[e];
  if (t0 >= cnt) return;
  int off_e = offsets[e];

  __shared__ __align__(16) __bf16 As[2][BM][LDK];
  __shared__ __align__(16) __bf16 Bs[2][BN][LDK];
  __shared__ int toks[BM];

  int tid = threadIdx.x;
  if (tid < BM) {
    int rank = t0 + tid;
    toks[tid] = (rank < cnt) ? token_list[e * Bt + rank] : 0;
  }
  __syncthreads();

  int lane = tid & 31, wave = tid >> 5;
  int moff = (wave & 1) * 64;
  int noff = (wave >> 1) * 32;

  // staging maps
  int ar = tid >> 1, ah = (tid & 1) * 16;       // A: row, 16-col half
  int bn = (tid & 31) * 4, bk = (tid >> 5) * 4; // B: 4n x 4k sub-block
  const float* aBase = x + (size_t)toks[ar] * Dd + ah;
  const float* W1e = W1 + (size_t)e * Dd * Hh;

  float4 aR[4], bR[4];
  auto gload = [&](int kt) {
    const float* asrc = aBase + kt;
#pragma unroll
    for (int i = 0; i < 4; ++i) aR[i] = ((const float4*)asrc)[i];
#pragma unroll
    for (int j = 0; j < 4; ++j)
      bR[j] = *(const float4*)(W1e + (size_t)(kt + bk + j) * Hh + n0 + bn);
  };
  auto sstore = [&](int buf) {
    __bf16* ad = &As[buf][ar][ah];
#pragma unroll
    for (int i = 0; i < 4; ++i) {
      const float* f = (const float*)&aR[i];
      ad[4 * i + 0] = (__bf16)f[0]; ad[4 * i + 1] = (__bf16)f[1];
      ad[4 * i + 2] = (__bf16)f[2]; ad[4 * i + 3] = (__bf16)f[3];
    }
#pragma unroll
    for (int i = 0; i < 4; ++i) {   // transpose 4x4 into [n][k]
      v4bf p;
      p[0] = (__bf16)((const float*)&bR[0])[i];
      p[1] = (__bf16)((const float*)&bR[1])[i];
      p[2] = (__bf16)((const float*)&bR[2])[i];
      p[3] = (__bf16)((const float*)&bR[3])[i];
      *(v4bf*)&Bs[buf][bn + i][bk] = p;
    }
  };

  v8f acc[4][2] = {};
  gload(0); sstore(0); __syncthreads();

  constexpr int NIT = Dd / BK;
  for (int it = 0; it < NIT; ++it) {
    int cur = it & 1;
    if (it + 1 < NIT) gload((it + 1) * BK);
#pragma unroll
    for (int mf = 0; mf < 4; ++mf) {
      v16bf a = load_afrag(&As[cur][0][0], moff + mf * 16, lane);
#pragma unroll
      for (int nf = 0; nf < 2; ++nf) {
        v16bf b = load_bfrag(&Bs[cur][0][0], noff + nf * 16, lane);
        acc[mf][nf] = __builtin_amdgcn_wmma_f32_16x16x32_bf16(
            false, a, false, b, (short)0, acc[mf][nf], false, false);
      }
    }
    if (it + 1 < NIT) sstore(cur ^ 1);
    __syncthreads();
  }

  // epilogue: bias + exact GELU -> bf16 h (compacted rows)
  int ln = lane & 15;
  int rhalf = (lane >> 4) * 8;
#pragma unroll
  for (int mf = 0; mf < 4; ++mf) {
#pragma unroll
    for (int nf = 0; nf < 2; ++nf) {
      int nabs = n0 + noff + nf * 16 + ln;
      float bias = b1[e * Hh + nabs];
#pragma unroll
      for (int r = 0; r < 8; ++r) {
        int m = moff + mf * 16 + rhalf + r;
        int rank = t0 + m;
        if (rank < cnt) {
          float v = acc[mf][nf][r] + bias;
          float g = 0.5f * v * (1.0f + erff(v * 0.70710678118f));
          h[(size_t)(off_e + rank) * Hh + nabs] = (__bf16)g;
        }
      }
    }
  }
}

// ---------------------------------------------------------------------------
// 4) fc2: y[token, n] += gate * ( h @ W2[e] + b2[e] )   (scatter atomicAdd)
//    A operand (h) is bf16 already -> async global->LDS copy when available.
// ---------------------------------------------------------------------------
__global__ __launch_bounds__(256) void fc2_kernel(
    const __bf16* __restrict__ h, const float* __restrict__ W2,
    const float* __restrict__ b2, const int* __restrict__ counts,
    const int* __restrict__ offsets, const int* __restrict__ token_list,
    const float* __restrict__ gate_vals, float* __restrict__ y) {
  int e  = blockIdx.z;
  int t0 = blockIdx.y * BM;
  int n0 = blockIdx.x * BN;
  int cnt = counts[e];
  if (t0 >= cnt) return;
  int off_e = offsets[e];

  __shared__ __align__(16) __bf16 As[2][BM][LDK];
  __shared__ __align__(16) __bf16 Bs[2][BN][LDK];
  __shared__ int toks[BM];
  __shared__ float gv[BM];

  int tid = threadIdx.x;
  if (tid < BM) {
    int rank = t0 + tid;
    toks[tid] = (rank < cnt) ? token_list[e * Bt + rank] : 0;
    gv[tid]   = (rank < cnt) ? gate_vals[e * Bt + rank] : 0.f;
  }
  __syncthreads();

  int lane = tid & 31, wave = tid >> 5;
  int moff = (wave & 1) * 64;
  int noff = (wave >> 1) * 32;

  // A staging: each thread owns row (tid>>1), two 16B chunks
  int ar = tid >> 1, ap = (tid & 1) * 16;
  int arank = t0 + ar;
  const __bf16* aBase =
      h + (size_t)(off_e + ((arank < cnt) ? arank : (cnt - 1))) * Hh + ap;
  // B staging: 4n x 4k sub-block
  int bn = (tid & 31) * 4, bk = (tid >> 5) * 4;
  const float* W2e = W2 + (size_t)e * Hh * Dd;

  float4 bR[4];
  auto gloadB = [&](int kt) {
#pragma unroll
    for (int j = 0; j < 4; ++j)
      bR[j] = *(const float4*)(W2e + (size_t)(kt + bk + j) * Dd + n0 + bn);
  };
  auto sstoreB = [&](int buf) {
#pragma unroll
    for (int i = 0; i < 4; ++i) {
      v4bf p;
      p[0] = (__bf16)((const float*)&bR[0])[i];
      p[1] = (__bf16)((const float*)&bR[1])[i];
      p[2] = (__bf16)((const float*)&bR[2])[i];
      p[3] = (__bf16)((const float*)&bR[3])[i];
      *(v4bf*)&Bs[buf][bn + i][bk] = p;
    }
  };

#if HAVE_ASYNC
  auto loadA_async = [&](int kt, int buf) {
    const __bf16* gsrc = aBase + kt;
    __builtin_amdgcn_global_load_async_to_lds_b128(
        (gv4i_t*)gsrc, (lv4i_t*)&As[buf][ar][ap], 0, 0);
    __builtin_amdgcn_global_load_async_to_lds_b128(
        (gv4i_t*)(gsrc + 8), (lv4i_t*)&As[buf][ar][ap + 8], 0, 0);
  };
#else
  v8bf aR[2];
  auto loadA_g = [&](int kt) {
    const __bf16* gsrc = aBase + kt;
    aR[0] = *(const v8bf*)(gsrc);
    aR[1] = *(const v8bf*)(gsrc + 8);
  };
  auto storeA_s = [&](int buf) {
    *(v8bf*)&As[buf][ar][ap]     = aR[0];
    *(v8bf*)&As[buf][ar][ap + 8] = aR[1];
  };
#endif

  v8f acc[4][2] = {};
  // prologue
#if HAVE_ASYNC
  loadA_async(0, 0);
  gloadB(0); sstoreB(0);
  wait_async0();
#else
  loadA_g(0); gloadB(0);
  storeA_s(0); sstoreB(0);
#endif
  __syncthreads();

  constexpr int NIT = Hh / BK;
  for (int it = 0; it < NIT; ++it) {
    int cur = it & 1;
    if (it + 1 < NIT) {
#if HAVE_ASYNC
      loadA_async((it + 1) * BK, cur ^ 1);
      gloadB((it + 1) * BK);
#else
      loadA_g((it + 1) * BK);
      gloadB((it + 1) * BK);
#endif
    }
#pragma unroll
    for (int mf = 0; mf < 4; ++mf) {
      v16bf a = load_afrag(&As[cur][0][0], moff + mf * 16, lane);
#pragma unroll
      for (int nf = 0; nf < 2; ++nf) {
        v16bf b = load_bfrag(&Bs[cur][0][0], noff + nf * 16, lane);
        acc[mf][nf] = __builtin_amdgcn_wmma_f32_16x16x32_bf16(
            false, a, false, b, (short)0, acc[mf][nf], false, false);
      }
    }
    if (it + 1 < NIT) {
#if HAVE_ASYNC
      sstoreB(cur ^ 1);
#else
      storeA_s(cur ^ 1); sstoreB(cur ^ 1);
#endif
    }
#if HAVE_ASYNC
    wait_async0();
#endif
    __syncthreads();
  }

  // epilogue: + b2, scale by gate, scatter-accumulate into y
  int ln = lane & 15;
  int rhalf = (lane >> 4) * 8;
#pragma unroll
  for (int mf = 0; mf < 4; ++mf) {
#pragma unroll
    for (int nf = 0; nf < 2; ++nf) {
      int nabs = n0 + noff + nf * 16 + ln;
      float bias = b2[e * Dd + nabs];
#pragma unroll
      for (int r = 0; r < 8; ++r) {
        int m = moff + mf * 16 + rhalf + r;
        int rank = t0 + m;
        if (rank < cnt) {
          float v = acc[mf][nf][r] + bias;
          atomicAdd(&y[(size_t)toks[m] * Dd + nabs], gv[m] * v);
        }
      }
    }
  }
}

// ---------------------------------------------------------------------------
// Launch
// ---------------------------------------------------------------------------
extern "C" void kernel_launch(void* const* d_in, const int* in_sizes, int n_in,
                              void* d_out, int out_size, void* d_ws, size_t ws_size,
                              hipStream_t stream) {
  const float* x      = (const float*)d_in[0];
  const float* w_gate = (const float*)d_in[1];
  const float* W1     = (const float*)d_in[2];
  const float* b1     = (const float*)d_in[3];
  const float* W2     = (const float*)d_in[4];
  const float* b2     = (const float*)d_in[5];
  float* y = (float*)d_out;

  // workspace layout
  char* ws = (char*)d_ws;
  int*   counts     = (int*)(ws + 0);           // 8 ints
  float* importance = (float*)(ws + 64);        // 8 floats
  int*   offsets    = (int*)(ws + 128);         // 8 ints
  int*   token_list = (int*)(ws + 256);                          // E*B ints (256 KB)
  float* gate_vals  = (float*)(ws + 256 + (size_t)Ee * Bt * 4);  // E*B floats (256 KB)
  __bf16* h         = (__bf16*)(ws + (size_t)(1 << 20));         // B*K x H bf16 (64 MB)

  (void)hipMemsetAsync(d_ws, 0, 256, stream);                               // counters
  (void)hipMemsetAsync(d_out, 0, (size_t)out_size * sizeof(float), stream); // y accum

  gate_kernel<<<Bt / 8, 256, 0, stream>>>(x, w_gate, counts, importance,
                                          token_list, gate_vals);
  finalize_kernel<<<1, 32, 0, stream>>>(counts, importance, offsets,
                                        y + (size_t)Bt * Dd);

  dim3 g1(Hh / BN, Bt / BM, Ee);  // 16 x 64 x 8
  fc1_kernel<<<g1, 256, 0, stream>>>(x, W1, b1, counts, offsets, token_list, h);

  dim3 g2(Dd / BN, Bt / BM, Ee);  // 8 x 64 x 8
  fc2_kernel<<<g2, 256, 0, stream>>>(h, W2, b2, counts, offsets, token_list,
                                     gate_vals, y);
}
